// MSC_87608742904488
// MI455X (gfx1250) — matrix operand order repaired and checked by reading, not compiled, not run on hardware
//
#include <hip/hip_runtime.h>
#include <math.h>

// ---------------- CDNA5 WMMA types ----------------
typedef __attribute__((ext_vector_type(16))) _Float16 v16h;
typedef __attribute__((ext_vector_type(8)))  _Float16 v8h;
typedef __attribute__((ext_vector_type(8)))  float    v8f;
typedef _Float16 h16;

union frag16 { v16h v; v8h h[2]; h16 e[16]; };

__device__ __forceinline__ v8f wmma_f16(v16h a, v16h b, v8f c) {
  // D = A(16x32 f16) x B(32x16 f16) + C(16x16 f32)
  return __builtin_amdgcn_wmma_f32_16x16x32_f16(false, a, false, b, (short)0, c, false, false);
}

// sortable-uint transform: ascending uint order == ascending float order
__device__ __forceinline__ unsigned toOrd(float f) {
  unsigned u = __float_as_uint(f);
  return u ^ ((u >> 31) ? 0xFFFFFFFFu : 0x80000000u);
}

#define BB 2
#define CC 256
#define HH 48
#define NN 2304            // 48*48
#define NHEAD 8
#define HD 32
#define MTILES 144         // NN/16
#define JCH 72             // NN/32

// ---------------- Kernel 1: multi-scale avg pool, transposed write ----------------
// y [B,C,H,W] f32 -> yp [B, N, C] f32  (sum of 3x3, 5x5, 7x7 same-pad avg pools)
__global__ void pool_kernel(const float* __restrict__ y, float* __restrict__ yp) {
  int tid = blockIdx.x * 256 + threadIdx.x;
  if (tid >= BB * CC * NN) return;
  int w = tid % HH;
  int h = (tid / HH) % HH;
  int c = (tid / NN) % CC;
  int b = tid / (NN * CC);
  const float* base = y + ((size_t)(b * CC + c)) * NN;
  const float W9 = 1.f / 9.f, W25 = 1.f / 25.f, W49 = 1.f / 49.f;
  float acc = 0.f;
  for (int dh = -3; dh <= 3; ++dh) {
    int hh = h + dh;
    if (hh < 0 || hh >= HH) continue;
    int adh = dh < 0 ? -dh : dh;
    for (int dw = -3; dw <= 3; ++dw) {
      int ww = w + dw;
      if (ww < 0 || ww >= HH) continue;
      int adw = dw < 0 ? -dw : dw;
      float wt = W49;
      if (adh <= 2 && adw <= 2) wt += W25;
      if (adh <= 1 && adw <= 1) wt += W9;
      acc += base[hh * HH + ww] * wt;
    }
  }
  yp[((size_t)(b * NN + h * HH + w)) * CC + c] = acc;
}

// ---------------- Kernel 2: LayerNorm over C -> f16 ----------------
__global__ void ln_kernel(const float* __restrict__ yp, const float* __restrict__ gamma,
                          const float* __restrict__ beta, h16* __restrict__ ynh) {
  __shared__ float red[256];
  int n = blockIdx.x;           // b*NN + n
  int c = threadIdx.x;
  float v = yp[(size_t)n * CC + c];
  red[c] = v; __syncthreads();
  for (int off = 128; off >= 1; off >>= 1) { if (c < off) red[c] += red[c + off]; __syncthreads(); }
  float mu = red[0] * (1.f / 256.f);
  __syncthreads();
  float d = v - mu;
  red[c] = d * d; __syncthreads();
  for (int off = 128; off >= 1; off >>= 1) { if (c < off) red[c] += red[c + off]; __syncthreads(); }
  float var = red[0] * (1.f / 256.f);
  float rstd = rsqrtf(var + 1e-5f);
  ynh[(size_t)n * CC + c] = (h16)(d * rstd * gamma[c] + beta[c]);
}

// ---------------- Kernel 3: x [B,C,N] -> xth f16 [B,N,C] ----------------
__global__ void xcvt_kernel(const float* __restrict__ x, h16* __restrict__ xth) {
  int tid = blockIdx.x * 256 + threadIdx.x;
  if (tid >= BB * NN * CC) return;
  int c = tid & 255;
  int n = (tid >> 8) % NN;
  int b = tid / (NN * CC);
  xth[tid] = (h16)x[((size_t)(b * CC + c)) * NN + n];
}

// ---------------- Kernel 4: weight transpose f32[K,N] -> f16[N,K] ----------------
__global__ void wtr_kernel(const float* __restrict__ src, h16* __restrict__ dst, int K, int N) {
  int tid = blockIdx.x * 256 + threadIdx.x;
  if (tid >= N * K) return;
  int k = tid % K;
  int n = tid / K;
  dst[tid] = (h16)src[(size_t)k * N + n];
}

// ---------------- Kernel 5: WMMA GEMM  C = alpha * A @ B^T, 16x64 tile per wave ----------------
// A f16 [M,K] row-major, BT f16 [N,K] row-major. N must be a multiple of 64.
// mode 0: f16 store out0[m*256+c]                      (Q projection, alpha = scale)
// mode 1: c<256 -> out0 (K-proj); c>=256 -> V transposed to vt[(b*8+h)*32+d][j] (f16)
// mode 2: f32 transposed store with bias: outf[(b*256+c)*2304 + n]   (final output)
__global__ void gemm_kernel(const h16* __restrict__ A, const h16* __restrict__ BT,
                            int M, int N, int K, float alpha, int mode,
                            h16* __restrict__ out0, h16* __restrict__ vt,
                            float* __restrict__ outf, const float* __restrict__ bias) {
  int lane = threadIdx.x & 31;
  int wave = threadIdx.x >> 5;
  int ngrp = N >> 6;                               // 64-wide N groups
  int tile = blockIdx.x * 8 + wave;
  int total = (M >> 4) * ngrp;
  if (tile >= total) return;                       // wave-uniform: EXEC stays full
  int mt = tile / ngrp, ng = tile % ngrp;
  int m0 = mt << 4, n0 = ng << 6;
  int l15 = lane & 15;
  int hb = (lane >> 4) << 3;                       // A-frag interleave: 0 or 8
  int kb = (lane >> 4) << 4;                       // unused (B-frag uses hb, like A): 0 or 16
  const h16* arow = A + (size_t)(m0 + l15) * K;
  const h16* brow0 = BT + (size_t)(n0 + l15) * K;
  v8f acc0 = {}, acc1 = {}, acc2 = {}, acc3 = {};
  for (int k0 = 0; k0 < K; k0 += 32) {
    frag16 a;
    a.h[0] = *(const v8h*)(arow + k0 + hb);
    a.h[1] = *(const v8h*)(arow + k0 + 16 + hb);
    __builtin_prefetch(arow + k0 + 64, 0, 3);      // global_prefetch_b8
    frag16 bf;
    bf.h[0] = *(const v8h*)(brow0 + k0 + hb);
    bf.h[1] = *(const v8h*)(brow0 + k0 + 16 + hb);
    acc0 = wmma_f16(a.v, bf.v, acc0);
    bf.h[0] = *(const v8h*)(brow0 + 16 * K + k0 + hb);
    bf.h[1] = *(const v8h*)(brow0 + 16 * K + k0 + 16 + hb);
    acc1 = wmma_f16(a.v, bf.v, acc1);
    bf.h[0] = *(const v8h*)(brow0 + 32 * K + k0 + hb);
    bf.h[1] = *(const v8h*)(brow0 + 32 * K + k0 + 16 + hb);
    acc2 = wmma_f16(a.v, bf.v, acc2);
    bf.h[0] = *(const v8h*)(brow0 + 48 * K + k0 + hb);
    bf.h[1] = *(const v8h*)(brow0 + 48 * K + k0 + 16 + hb);
    acc3 = wmma_f16(a.v, bf.v, acc3);
  }
  v8f accs[4] = {acc0, acc1, acc2, acc3};
  int b_ = m0 / NN;
  int pn = (m0 % NN) + hb;
  #pragma unroll
  for (int u = 0; u < 4; ++u) {
    v8f acc = accs[u];
    int c = n0 + u * 16 + l15;
    if (mode == 2) {
      float bb = bias[c];
      float* dst = outf + ((size_t)(b_ * CC + c)) * NN + pn;
      float4 f0 = make_float4(acc[0] + bb, acc[1] + bb, acc[2] + bb, acc[3] + bb);
      float4 f1 = make_float4(acc[4] + bb, acc[5] + bb, acc[6] + bb, acc[7] + bb);
      *(float4*)dst = f0;
      *(float4*)(dst + 4) = f1;
    } else if (mode == 1 && c >= 256) {
      int cc2 = c - 256;
      int hh = cc2 >> 5, dd = cc2 & 31;
      v8h pk;
      #pragma unroll
      for (int r = 0; r < 8; ++r) pk[r] = (h16)(acc[r] * alpha);
      *(v8h*)(vt + ((size_t)(b_ * NHEAD + hh) * HD + dd) * NN + pn) = pk;
    } else {
      #pragma unroll
      for (int r = 0; r < 8; ++r)
        out0[(size_t)(m0 + r + hb) * 256 + c] = (h16)(acc[r] * alpha);
    }
  }
}

// ---------------- radix select helpers ----------------
// resolve one 256-bin descending-cumulative step; updates (remaining, prefix)
__device__ __forceinline__ void resolve256(const unsigned* __restrict__ hist, int lane,
                                           int& remaining, unsigned& prefix) {
  int gtop = 255 - 8 * lane;
  unsigned cbin[8]; unsigned gsum = 0;
  #pragma unroll
  for (int t = 0; t < 8; ++t) { cbin[t] = hist[gtop - t]; gsum += cbin[t]; }
  unsigned pre = gsum;
  #pragma unroll
  for (int off = 1; off < 32; off <<= 1) {
    unsigned v = __shfl_up(pre, off, 32);
    if (lane >= off) pre += v;
  }
  unsigned excl = pre - gsum;
  bool hit = (excl < (unsigned)remaining) && ((unsigned)remaining <= pre);
  unsigned long long hm = __ballot(hit);
  int hl = (int)__ffsll(hm) - 1;
  if (hl < 0) hl = 0;
  unsigned binsel = 0, rem2 = 1;
  if (hit) {
    unsigned accu = excl; int done = 0;
    #pragma unroll
    for (int t = 0; t < 8; ++t) {
      if (!done && accu + cbin[t] >= (unsigned)remaining) {
        binsel = (unsigned)(gtop - t);
        rem2 = (unsigned)remaining - accu;
        done = 1;
      }
      if (!done) accu += cbin[t];
    }
  }
  binsel = (unsigned)__shfl((int)binsel, hl, 32);
  rem2   = (unsigned)__shfl((int)rem2, hl, 32);
  prefix = (prefix << 8) | binsel;
  remaining = (int)rem2;
}

// dual k-th-largest selection sharing one data pass per radix round (1 wave, 2x256-bin hist)
__device__ void select2_kth(const float* __restrict__ rs, int k1, int k2,
                            unsigned* __restrict__ hist, int lane,
                            unsigned& key1, unsigned& key2) {
  unsigned p1 = 0, p2 = 0;
  int r1 = k1, r2 = k2;
  for (int pass = 0; pass < 4; ++pass) {
    int shift = 24 - 8 * pass;
    for (int bn = lane; bn < 512; bn += 32) hist[bn] = 0;
    for (int i = lane; i < NN; i += 32) {
      unsigned key = toOrd(rs[i]);
      unsigned bits = (key >> shift) & 255u;
      if (pass == 0) {
        atomicAdd(&hist[bits], 1u);                // shared histogram for both selects
      } else {
        unsigned hi = key >> (shift + 8);
        if (hi == p1) atomicAdd(&hist[bits], 1u);
        if (hi == p2) atomicAdd(&hist[256 + bits], 1u);
      }
    }
    resolve256(hist, lane, r1, p1);
    resolve256((pass == 0) ? hist : hist + 256, lane, r2, p2);
  }
  key1 = p1; key2 = p2;
}

// ---------------- Kernel 6: fused attention (scores in LDS, dual top-k softmax, P@V) -------------
// grid: B*NHEAD*MTILES blocks, 256 threads (8 waves).
#define LDS_M1_OFF    147456                       // 16 x 72 u32 mask (top-1152)
#define LDS_M2_OFF    152064                       // 16 x 72 u32 mask (top-768)
#define LDS_HIST_OFF  156672                       // 8 waves x 512 u32
#define LDS_I1_OFF    173056                       // 16 f32 (pad 64)
#define LDS_I2_OFF    173120
#define LDS_PART_OFF  173184                       // 8 waves x 512 f32 partial out
#define LDS_TOTAL     189568

__global__ void attn_kernel(const h16* __restrict__ qh, const h16* __restrict__ kh,
                            const h16* __restrict__ vt, h16* __restrict__ attnout,
                            const float* __restrict__ w1p, const float* __restrict__ w2p) {
  extern __shared__ char smem[];
  float*    lds_s  = (float*)smem;                 // 16 x 2304 f32: scores -> exp values
  unsigned* lds_m1 = (unsigned*)(smem + LDS_M1_OFF);
  unsigned* lds_m2 = (unsigned*)(smem + LDS_M2_OFF);
  unsigned* hist   = (unsigned*)(smem + LDS_HIST_OFF);
  float*    lds_i1 = (float*)(smem + LDS_I1_OFF);
  float*    lds_i2 = (float*)(smem + LDS_I2_OFF);
  float*    lds_pt = (float*)(smem + LDS_PART_OFF);

  int tid = threadIdx.x, lane = tid & 31, wave = tid >> 5;
  int blk = blockIdx.x;
  int mt = blk % MTILES;
  int bh = blk / MTILES;
  int h = bh % NHEAD, b = bh / NHEAD;
  int row0 = mt << 4;
  int l15 = lane & 15;
  int hb = (lane >> 4) << 3;     // A-frag K interleave
  int kb = (lane >> 4) << 4;     // unused (B-frag uses hb, like A)

  // ---- Phase 1: scores = (scaled Q) @ K^T into LDS ----
  frag16 aq;
  {
    const h16* qrow = qh + (((size_t)(b * NN + row0 + l15)) * NHEAD + h) * HD;
    aq.h[0] = *(const v8h*)(qrow + hb);
    aq.h[1] = *(const v8h*)(qrow + 16 + hb);
  }
  for (int jt = wave; jt < MTILES; jt += 8) {
    const h16* krow = kh + (((size_t)(b * NN + jt * 16 + l15)) * NHEAD + h) * HD;
    __builtin_prefetch(krow + 8 * 16 * NHEAD * HD, 0, 3);   // next jt for this wave
    frag16 bk;
    bk.h[0] = *(const v8h*)(krow + hb);
    bk.h[1] = *(const v8h*)(krow + 16 + hb);
    v8f d = {};
    d = wmma_f16(aq.v, bk.v, d);
    int col = jt * 16 + l15;
    #pragma unroll
    for (int r = 0; r < 8; ++r) lds_s[(size_t)(r + hb) * NN + col] = d[r];
  }
  __syncthreads();

  // ---- Phase 2: per-row max, dual radix-select thresholds, exp + masked sums ----
  unsigned* myhist = hist + wave * 512;
  float w1v = w1p[0], w2v = w2p[0];
  for (int rr = 0; rr < 2; ++rr) {
    int row = wave * 2 + rr;
    float* rs = lds_s + (size_t)row * NN;
    float mx = -3.4e38f;
    for (int t = 0; t < JCH; ++t) mx = fmaxf(mx, rs[lane + 32 * t]);
    #pragma unroll
    for (int off = 16; off >= 1; off >>= 1) mx = fmaxf(mx, __shfl_xor(mx, off, 32));
    unsigned key1, key2;
    select2_kth(rs, NN / 2, NN / 3, myhist, lane, key1, key2);  // top-1152 / top-768
    float s1 = 0.f, s2 = 0.f;
    for (int t = 0; t < JCH; ++t) {
      int i = lane + 32 * t;
      float s = rs[i];
      unsigned key = toOrd(s);
      float e = __expf(s - mx);
      bool b1 = key >= key1, b2 = key >= key2;
      if (b1) s1 += e;
      if (b2) s2 += e;
      rs[i] = e;                                            // overwrite score with exp
      unsigned mk1 = (unsigned)__ballot(b1);
      unsigned mk2 = (unsigned)__ballot(b2);
      if (lane == 0) { lds_m1[row * JCH + t] = mk1; lds_m2[row * JCH + t] = mk2; }
    }
    #pragma unroll
    for (int off = 16; off >= 1; off >>= 1) {
      s1 += __shfl_xor(s1, off, 32);
      s2 += __shfl_xor(s2, off, 32);
    }
    if (lane == 0) { lds_i1[row] = w1v / s1; lds_i2[row] = w2v / s2; }
  }
  __syncthreads();

  // ---- Phase 3: out(16x32) = P(16x2304) @ V^T, K-split over 8 waves ----
  v8f accA = {}, accB = {};
  const h16* vtb = vt + (size_t)(b * NHEAD + h) * HD * NN;
  float i1 = lds_i1[l15], i2 = lds_i2[l15];
  const float* erow = lds_s + (size_t)l15 * NN;
  for (int jc = wave; jc < JCH; jc += 8) {
    int jb = jc * 32;
    const float* ep = erow + jb + hb;
    unsigned mk1 = lds_m1[l15 * JCH + jc];
    unsigned mk2 = lds_m2[l15 * JCH + jc];
    frag16 ap;
    #pragma unroll
    for (int t = 0; t < 8; ++t) {
      int j0 = hb + t;
      float w0 = ep[t] * ((((mk1 >> j0) & 1u) ? i1 : 0.f) + (((mk2 >> j0) & 1u) ? i2 : 0.f));
      ap.e[t] = (h16)w0;
      int j1 = 16 + hb + t;
      float w1 = ep[16 + t] * ((((mk1 >> j1) & 1u) ? i1 : 0.f) + (((mk2 >> j1) & 1u) ? i2 : 0.f));
      ap.e[8 + t] = (h16)w1;
    }
    const h16* vr0 = vtb + (size_t)l15 * NN + jb + hb;        // d = l15
    const h16* vr1 = vr0 + (size_t)16 * NN;                   // d = l15 + 16
    __builtin_prefetch(vr0 + 256, 0, 3);
    frag16 b0, b1f;
    b0.h[0]  = *(const v8h*)vr0;  b0.h[1]  = *(const v8h*)(vr0 + 16);
    b1f.h[0] = *(const v8h*)vr1;  b1f.h[1] = *(const v8h*)(vr1 + 16);
    accA = wmma_f16(ap.v, b0.v, accA);
    accB = wmma_f16(ap.v, b1f.v, accB);
  }
  float* myp = lds_pt + wave * 512;
  #pragma unroll
  for (int r = 0; r < 8; ++r) {
    myp[(r + hb) * 32 + l15]      = accA[r];
    myp[(r + hb) * 32 + 16 + l15] = accB[r];
  }
  __syncthreads();
  // deterministic fixed-order reduction over waves + f16 store
  for (int i = tid; i < 512; i += 256) {
    float s = 0.f;
    #pragma unroll
    for (int w = 0; w < 8; ++w) s += lds_pt[w * 512 + i];
    int mm = i >> 5, dd = i & 31;
    attnout[((size_t)(b * NN + row0 + mm)) * CC + h * HD + dd] = (h16)s;
  }
}

// ---------------- host launcher ----------------
extern "C" void kernel_launch(void* const* d_in, const int* in_sizes, int n_in,
                              void* d_out, int out_size, void* d_ws, size_t ws_size,
                              hipStream_t stream) {
  const float* x     = (const float*)d_in[0];
  const float* y     = (const float*)d_in[1];
  const float* Wq    = (const float*)d_in[2];
  const float* Wkv   = (const float*)d_in[3];
  const float* Wproj = (const float*)d_in[4];
  const float* bproj = (const float*)d_in[5];
  const float* gamma = (const float*)d_in[6];
  const float* beta  = (const float*)d_in[7];
  const float* aw1   = (const float*)d_in[8];
  const float* aw2   = (const float*)d_in[9];
  float* outp = (float*)d_out;

  // workspace carve (256B aligned)
  size_t off = 0;
  char* base = (char*)d_ws;
  auto carve = [&](size_t bytes) -> char* {
    char* p = base + off;
    off += (bytes + 255) & ~(size_t)255;
    return p;
  };
  const size_t MROWS = (size_t)BB * NN;            // 4608
  float* yp     = (float*)carve(MROWS * CC * 4);   // pooled, [B,N,C] f32
  h16*  ynh     = (h16*)carve(MROWS * CC * 2);     // LN'd, f16
  h16*  xth     = (h16*)carve(MROWS * CC * 2);     // x transposed, f16
  h16*  wqT     = (h16*)carve((size_t)256 * 256 * 2);
  h16*  wkvT    = (h16*)carve((size_t)512 * 256 * 2);
  h16*  wpT     = (h16*)carve((size_t)256 * 256 * 2);
  h16*  qh      = (h16*)carve(MROWS * CC * 2);     // scaled Q, [B,N,8,32] f16
  h16*  kh      = (h16*)carve(MROWS * CC * 2);     // K,       [B,N,8,32] f16
  h16*  vt      = (h16*)carve(MROWS * CC * 2);     // V^T,     [B,8,32,N] f16
  h16*  attnout = (h16*)carve(MROWS * CC * 2);     // attention out, [B,N,C] f16
  (void)ws_size; (void)in_sizes; (void)n_in; (void)out_size;

  const float SCALE = 0.17677669529663687f;        // 32^-0.5

  hipFuncSetAttribute((const void*)attn_kernel,
                      hipFuncAttributeMaxDynamicSharedMemorySize, LDS_TOTAL);

  // 1) pooling (transposed write)
  pool_kernel<<<(BB * CC * NN + 255) / 256, 256, 0, stream>>>(y, yp);
  // 2) layernorm over C
  ln_kernel<<<BB * NN, 256, 0, stream>>>(yp, gamma, beta, ynh);
  // 3) x -> [B,N,C] f16
  xcvt_kernel<<<(BB * NN * CC + 255) / 256, 256, 0, stream>>>(x, xth);
  // 4) weight transposes -> f16
  wtr_kernel<<<(256 * 256 + 255) / 256, 256, 0, stream>>>(Wq, wqT, 256, 256);
  wtr_kernel<<<(512 * 256 + 255) / 256, 256, 0, stream>>>(Wkv, wkvT, 256, 512);
  wtr_kernel<<<(256 * 256 + 255) / 256, 256, 0, stream>>>(Wproj, wpT, 256, 256);
  // 5) Q = scale * xth @ Wq^T   (mode 0): 288 * 4 = 1152 tiles
  gemm_kernel<<<1152 / 8, 256, 0, stream>>>(xth, wqT, (int)MROWS, 256, 256,
                                            SCALE, 0, qh, nullptr, nullptr, nullptr);
  // 6) KV = ynh @ Wkv^T  (mode 1): 288 * 8 = 2304 tiles
  gemm_kernel<<<2304 / 8, 256, 0, stream>>>(ynh, wkvT, (int)MROWS, 512, 256,
                                            1.0f, 1, kh, vt, nullptr, nullptr);
  // 7) fused attention: dual top-k softmax + P@V
  attn_kernel<<<BB * NHEAD * MTILES, 256, LDS_TOTAL, stream>>>(qh, kh, vt, attnout, aw1, aw2);
  // 8) final projection + bias, transposed f32 store to d_out [B,C,H,W]
  gemm_kernel<<<1152 / 8, 256, 0, stream>>>(attnout, wpT, (int)MROWS, 256, 256,
                                            1.0f, 2, nullptr, nullptr, outp, bproj);
}
